// BiUpsampler_54829552500771
// MI455X (gfx1250) — compile-verified, matrix operand order checked by reading, not run on hardware
//
#include <hip/hip_runtime.h>
#include <hip/hip_bf16.h>

typedef __attribute__((ext_vector_type(16))) __bf16 bf16x16;
typedef __attribute__((ext_vector_type(8)))  float  floatx8;
typedef __attribute__((ext_vector_type(4)))  unsigned int uint32x4;
typedef __attribute__((ext_vector_type(4)))  int    int32x4;
typedef __attribute__((ext_vector_type(8)))  int    int32x8;

#define HO 192
#define WO 192
#define P  (HO * WO)   // 36864
#define HI 96
#define WI 96
#define NC 64          // feature channels C
#define BS 2

// K index inside a 32-wide K-chunk for the ISA 16-bit A-matrix layout:
// lane group kb (0: lanes 0-15, 1: lanes 16-31), vector element e (0..15).
// VGPR j = e>>1 holds K = {0 or 16} + kb*8 + (j&3)*2 + (e&1).
__device__ __forceinline__ int kmap(int e, int kb) {
    int j = e >> 1;
    return ((j & 4) ? 16 : 0) + (kb << 3) + ((j & 3) << 1) + (e & 1);
}

// -------------------- head: 1x1 conv 3 -> 64, relu, both branches ----------
__global__ void head_kernel(const float* __restrict__ pose,
                            const float* __restrict__ dw_w1, const float* __restrict__ dw_b1,
                            const float* __restrict__ pw_w1, const float* __restrict__ pw_b1,
                            __bf16* __restrict__ f1dw, __bf16* __restrict__ f1pw)
{
    int idx = blockIdx.x * blockDim.x + threadIdx.x;
    if (idx >= NC * P) return;
    int oc = idx / P, p = idx - oc * P;
    float p0 = pose[p], p1 = pose[P + p], p2 = pose[2 * P + p];
    float a = dw_b1[oc] + dw_w1[oc * 3] * p0 + dw_w1[oc * 3 + 1] * p1 + dw_w1[oc * 3 + 2] * p2;
    float b = pw_b1[oc] + pw_w1[oc * 3] * p0 + pw_w1[oc * 3 + 1] * p1 + pw_w1[oc * 3 + 2] * p2;
    f1dw[idx] = (__bf16)fmaxf(a, 0.f);
    f1pw[idx] = (__bf16)fmaxf(b, 0.f);
}

// -------------------- 1x1 conv as WMMA GEMM with TDM-staged B panel --------
// Block = 256 threads = 8 waves; computes a 16(oc) x 128(pixel) tile.
// Wave 0 issues one tensor_load_to_lds for the whole KTOT x 128 bf16 B panel
// (2-D tile: tile_dim0 = 128 px, tile_dim1 = KTOT rows, row stride = P).
// TDM pad insertion adds 1 DWORD after each 256B row -> LDS row = 130 bf16.
template<int KTOT, bool RELU, bool WF32, bool WBF16>
__global__ void gemm1x1_kernel(const float* __restrict__ W, const float* __restrict__ bias,
                               const __bf16* __restrict__ In,
                               float* __restrict__ OutF, __bf16* __restrict__ OutB)
{
    constexpr int LROW = 130;                  // 128 data + 1 DWORD TDM pad (bank skew)
    __shared__ __bf16 ldsB[KTOT * LROW];

    const int lane   = threadIdx.x & 31;
    const int wave   = threadIdx.x >> 5;
    const int pBlock = blockIdx.x << 7;        // 128 pixels per block
    const int ocBase = blockIdx.y << 4;        // 16 out-channels
    const int mn = lane & 15;
    const int kb = (lane >> 4) & 1;

    if (threadIdx.x == 0) {
        const unsigned long long ga = (unsigned long long)(size_t)(In + pBlock);
        const unsigned ldsOff = (unsigned)(size_t)&ldsB[0];   // low 32b of flat = LDS offset
        uint32x4 g0;
        g0[0] = 1u;                                           // count = 1 (valid D#)
        g0[1] = ldsOff;                                       // lds_addr
        g0[2] = (unsigned)(ga & 0xffffffffull);               // global_addr[31:0]
        g0[3] = (unsigned)((ga >> 32) & 0x01ffffffull) | (2u << 30); // addr[56:32] | type=2
        int32x8 g1;
        g1[0] = (int)(0x00010000u | (1u << 20) | (5u << 22)); // data_size=2B | pad_en | 64DW interval, 1DW pad
        g1[1] = (int)(((unsigned)P & 0xffffu) << 16);         // tensor_dim0[15:0]
        g1[2] = (int)(((unsigned)KTOT & 0xffffu) << 16);      // tensor_dim0[31:16]=0 | tensor_dim1[15:0]
        g1[3] = (int)(128u << 16);                            // tensor_dim1[31:16]=0 | tile_dim0=128
        g1[4] = (int)KTOT;                                    // tile_dim1=KTOT | tile_dim2=0
        g1[5] = (int)P;                                       // tensor_dim0_stride[31:0]
        g1[6] = 0;                                            // stride hi | dim1_stride lo
        g1[7] = 0;                                            // dim1_stride hi
        int32x4 z4 = {};
#if defined(__clang_major__) && (__clang_major__ >= 23)
        int32x8 z8 = {};
        __builtin_amdgcn_tensor_load_to_lds(g0, g1, z4, z4, z8, 0);
#else
        __builtin_amdgcn_tensor_load_to_lds(g0, g1, z4, z4, 0);
#endif
        __builtin_amdgcn_s_wait_tensorcnt(0);
    }
    __syncthreads();

    const int col = (wave << 4) + mn;
    floatx8 acc = {};
    for (int k0 = 0; k0 < KTOT; k0 += 32) {
        bf16x16 a, b;
#pragma unroll
        for (int e = 0; e < 16; ++e) {
            const int k = k0 + kmap(e, kb);
            a[e] = (__bf16)W[(size_t)(ocBase + mn) * KTOT + k];  // fp32 wts -> bf16
            b[e] = ldsB[k * LROW + col];
        }
        acc = __builtin_amdgcn_wmma_f32_16x16x32_bf16(false, a, false, b,
                                                      (short)0, acc, false, false);
    }
#pragma unroll
    for (int r = 0; r < 8; ++r) {
        const int m = ocBase + r + (kb << 3);
        float v = acc[r] + bias[m];
        if constexpr (RELU) v = fmaxf(v, 0.f);
        const size_t o = (size_t)m * P + pBlock + col;
        if constexpr (WF32)  OutF[o] = v;
        if constexpr (WBF16) OutB[o] = (__bf16)v;
    }
}

// -------------------- 3x3 conv (pad 1) as implicit WMMA GEMM, K = 64*9 -----
template<int OCV, int OCTOT, bool RELU, bool WF32, bool WBF16>
__global__ void conv3x3_kernel(const float* __restrict__ W,   // [OCTOT][64*9] row-major
                               const float* __restrict__ bias,
                               const __bf16* __restrict__ In, // [BS][64][HO][WO]
                               float* __restrict__ OutF, __bf16* __restrict__ OutB)
{
    const int lane  = threadIdx.x & 31;
    const int wave  = threadIdx.x >> 5;
    const int pTile = (blockIdx.x * (blockDim.x >> 5) + wave) << 4;
    const int ocBase = blockIdx.y << 4;
    const int bIdx  = blockIdx.z;
    const int y  = pTile / WO;          // tile never crosses a row (192 % 16 == 0)
    const int x0 = pTile - y * WO;
    const int mn = lane & 15;
    const int kb = (lane >> 4) & 1;
    const __bf16* inB = In + (size_t)bIdx * NC * P;

    floatx8 acc = {};
    for (int k0 = 0; k0 < NC * 9; k0 += 32) {
        bf16x16 a, b;
#pragma unroll
        for (int e = 0; e < 16; ++e) {
            const int k = k0 + kmap(e, kb);
            a[e] = (OCV == 16 || mn < OCV) ? (__bf16)W[(size_t)(ocBase + mn) * (NC * 9) + k]
                                           : (__bf16)0.f;
            const int ic = k / 9, r9 = k - ic * 9;
            const int yy = y + r9 / 3 - 1;
            const int xx = x0 + mn + (r9 % 3) - 1;
            b[e] = (yy >= 0 && yy < HO && xx >= 0 && xx < WO)
                   ? inB[((size_t)ic * HO + yy) * WO + xx] : (__bf16)0.f;
        }
        acc = __builtin_amdgcn_wmma_f32_16x16x32_bf16(false, a, false, b,
                                                      (short)0, acc, false, false);
    }
#pragma unroll
    for (int r = 0; r < 8; ++r) {
        const int m = ocBase + r + (kb << 3);
        if (OCV == 16 || m < OCV) {
            float v = acc[r] + bias[m];
            if constexpr (RELU) v = fmaxf(v, 0.f);
            const size_t o = ((size_t)bIdx * OCTOT + m) * P + pTile + mn;
            if constexpr (WF32)  OutF[o] = v;
            if constexpr (WBF16) OutB[o] = (__bf16)v;
        }
    }
}

// -------------------- mean = sum_k h * softmax(dw) -------------------------
__global__ void mean_kernel(const float* __restrict__ x, const float* __restrict__ dwf,
                            const int* __restrict__ iy, const int* __restrict__ ix,
                            float* __restrict__ meanF, __bf16* __restrict__ meanB)
{
    int idx = blockIdx.x * blockDim.x + threadIdx.x;
    if (idx >= BS * NC * P) return;
    int p = idx % P;
    int c = (idx / P) % NC;
    int b = idx / (NC * P);
    int Y = iy[p], X = ix[p];
    float h[9], w[9];
    float mx = -1e30f;
#pragma unroll
    for (int k = 0; k < 9; ++k) {
        int yy = Y + k / 3 - 1, xx = X + k % 3 - 1;
        h[k] = (yy >= 0 && yy < HI && xx >= 0 && xx < WI)
               ? x[(((size_t)b * NC + c) * HI + yy) * WI + xx] : 0.f;
        w[k] = dwf[(size_t)(c * 9 + k) * P + p];
        mx = fmaxf(mx, w[k]);
    }
    float s = 0.f, acc = 0.f;
#pragma unroll
    for (int k = 0; k < 9; ++k) {
        float e = __expf(w[k] - mx);
        s += e;
        acc += h[k] * e;
    }
    float mv = acc / s;
    meanF[idx] = mv;
    meanB[idx] = (__bf16)mv;
}

// -------------------- xa = sum_k h * softmax(sd*dw + sr*|h-mean|) ----------
__global__ void xa_kernel(const float* __restrict__ x, const float* __restrict__ dwf,
                          const int* __restrict__ iy, const int* __restrict__ ix,
                          const float* __restrict__ meanF, const float* __restrict__ sigmaF,
                          float* __restrict__ xa)
{
    int idx = blockIdx.x * blockDim.x + threadIdx.x;
    if (idx >= BS * NC * P) return;
    int p = idx % P;
    int c = (idx / P) % NC;
    int b = idx / (NC * P);
    int Y = iy[p], X = ix[p];
    float sd = sigmaF[((size_t)b * 2 + 0) * P + p];
    float sr = sigmaF[((size_t)b * 2 + 1) * P + p];
    float mv = meanF[idx];
    float h[9], bw[9];
    float mx = -1e30f;
#pragma unroll
    for (int k = 0; k < 9; ++k) {
        int yy = Y + k / 3 - 1, xx = X + k % 3 - 1;
        h[k] = (yy >= 0 && yy < HI && xx >= 0 && xx < WI)
               ? x[(((size_t)b * NC + c) * HI + yy) * WI + xx] : 0.f;
        float dwv = dwf[(size_t)(c * 9 + k) * P + p];
        bw[k] = sd * dwv + sr * fabsf(h[k] - mv);
        mx = fmaxf(mx, bw[k]);
    }
    float s = 0.f, acc = 0.f;
#pragma unroll
    for (int k = 0; k < 9; ++k) {
        float e = __expf(bw[k] - mx);
        s += e;
        acc += h[k] * e;
    }
    xa[idx] = acc / s;
}

// -------------------- out[b,o,p] = sum_c xa[b,c,p] * pw[c*3+o, p] ----------
__global__ void out_kernel(const float* __restrict__ xa, const float* __restrict__ pwf,
                           float* __restrict__ out)
{
    int idx = blockIdx.x * blockDim.x + threadIdx.x;
    if (idx >= BS * 3 * P) return;
    int p = idx % P;
    int o = (idx / P) % 3;
    int b = idx / (3 * P);
    float s = 0.f;
#pragma unroll 4
    for (int c = 0; c < NC; ++c)
        s += xa[((size_t)b * NC + c) * P + p] * pwf[(size_t)(c * 3 + o) * P + p];
    out[idx] = s;
}

extern "C" void kernel_launch(void* const* d_in, const int* in_sizes, int n_in,
                              void* d_out, int out_size, void* d_ws, size_t ws_size,
                              hipStream_t stream)
{
    (void)in_sizes; (void)n_in; (void)out_size; (void)ws_size;
    const float* x      = (const float*)d_in[0];
    const float* pose   = (const float*)d_in[1];
    const int*   iy     = (const int*)d_in[2];
    const int*   ix     = (const int*)d_in[3];
    const float* sig_w1 = (const float*)d_in[4];
    const float* sig_b1 = (const float*)d_in[5];
    const float* sig_w2 = (const float*)d_in[6];
    const float* sig_b2 = (const float*)d_in[7];
    const float* sig_w3 = (const float*)d_in[8];
    const float* sig_b3 = (const float*)d_in[9];
    const float* dw_w1  = (const float*)d_in[10];
    const float* dw_b1  = (const float*)d_in[11];
    const float* dw_w2  = (const float*)d_in[12];
    const float* dw_b2  = (const float*)d_in[13];
    const float* dw_w3  = (const float*)d_in[14];
    const float* dw_b3  = (const float*)d_in[15];
    const float* pw_w1  = (const float*)d_in[16];
    const float* pw_b1  = (const float*)d_in[17];
    const float* pw_w2  = (const float*)d_in[18];
    const float* pw_b2  = (const float*)d_in[19];
    const float* pw_w3  = (const float*)d_in[20];
    const float* pw_b3  = (const float*)d_in[21];
    float* out = (float*)d_out;

    char* ws = (char*)d_ws;
    auto take = [&](size_t bytes) { char* r = ws; ws += (bytes + 255) & ~(size_t)255; return r; };

    __bf16* f1dw  = (__bf16*)take((size_t)NC * P * sizeof(__bf16));
    __bf16* f1pw  = (__bf16*)take((size_t)NC * P * sizeof(__bf16));
    __bf16* f2dw  = (__bf16*)take((size_t)256 * P * sizeof(__bf16));
    __bf16* f2pw  = (__bf16*)take((size_t)256 * P * sizeof(__bf16));
    float*  dwf   = (float*)take((size_t)576 * P * sizeof(float));
    float*  pwf   = (float*)take((size_t)192 * P * sizeof(float));
    float*  meanF = (float*)take((size_t)BS * NC * P * sizeof(float));
    __bf16* meanB = (__bf16*)take((size_t)BS * NC * P * sizeof(__bf16));
    __bf16* s1    = (__bf16*)take((size_t)BS * NC * P * sizeof(__bf16));
    __bf16* s2    = (__bf16*)take((size_t)BS * NC * P * sizeof(__bf16));
    float*  sigF  = (float*)take((size_t)BS * 2 * P * sizeof(float));
    float*  xa    = (float*)take((size_t)BS * NC * P * sizeof(float));

    const int TPB = 256;
    const int PBLK_X = P / 128;                  // 288, exact (TDM GEMM blocks)
    const int PTILES_X = P / (16 * (TPB / 32));  // 288, exact (conv blocks)

    // 1) head 3->64 (both branches)
    head_kernel<<<(NC * P + TPB - 1) / TPB, TPB, 0, stream>>>(
        pose, dw_w1, dw_b1, pw_w1, pw_b1, f1dw, f1pw);

    // 2) WMMA GEMM 64->256 + relu (bf16 out), TDM-staged B, both branches
    gemm1x1_kernel<64, true, false, true><<<dim3(PBLK_X, 256 / 16), TPB, 0, stream>>>(
        dw_w2, dw_b2, f1dw, nullptr, f2dw);
    gemm1x1_kernel<64, true, false, true><<<dim3(PBLK_X, 256 / 16), TPB, 0, stream>>>(
        pw_w2, pw_b2, f1pw, nullptr, f2pw);

    // 3) WMMA GEMM 256->576 (dw) and 256->192 (pw), f32 out, TDM-staged B
    gemm1x1_kernel<256, false, true, false><<<dim3(PBLK_X, 576 / 16), TPB, 0, stream>>>(
        dw_w3, dw_b3, f2dw, dwf, nullptr);
    gemm1x1_kernel<256, false, true, false><<<dim3(PBLK_X, 192 / 16), TPB, 0, stream>>>(
        pw_w3, pw_b3, f2pw, pwf, nullptr);

    // 4) gather + softmax(dw) -> mean
    mean_kernel<<<(BS * NC * P + TPB - 1) / TPB, TPB, 0, stream>>>(
        x, dwf, iy, ix, meanF, meanB);

    // 5) sigma chain: 3x3 implicit-GEMM WMMA convs
    conv3x3_kernel<16, 64, true, false, true><<<dim3(PTILES_X, 4, BS), TPB, 0, stream>>>(
        sig_w1, sig_b1, meanB, nullptr, s1);
    conv3x3_kernel<16, 64, true, false, true><<<dim3(PTILES_X, 4, BS), TPB, 0, stream>>>(
        sig_w2, sig_b2, s1, nullptr, s2);
    conv3x3_kernel<2, 2, false, true, false><<<dim3(PTILES_X, 1, BS), TPB, 0, stream>>>(
        sig_w3, sig_b3, s2, sigF, nullptr);

    // 6) xa = sum_k h * softmax(bw)
    xa_kernel<<<(BS * NC * P + TPB - 1) / TPB, TPB, 0, stream>>>(
        x, dwf, iy, ix, meanF, sigF, xa);

    // 7) out[b,o,p] = sum_c xa[b,c,p] * pw[c,o,p]
    out_kernel<<<(BS * 3 * P + TPB - 1) / TPB, TPB, 0, stream>>>(xa, pwf, out);
}
